// Trainer_GCN_23450521436540
// MI455X (gfx1250) — matrix operand order repaired and checked by reading, not compiled, not run on hardware
//
#include <hip/hip_runtime.h>

// ---------------------------------------------------------------------------
// GAT (2x GATConv, heads=1, edge_dim=1) forward for MI455X / gfx1250.
// GEMMs: wave32 v_wmma_f32_16x16x32_f16, A streamed from global (b128),
// B staged per-block into LDS via async-to-LDS (double buffered, ASYNCcnt).
// Scatter softmax + aggregation: f32 global atomics (bandwidth bound:
// ~3.3 GB of gather/scatter -> ~140us @ 23.3 TB/s; GEMMs are negligible).
// ---------------------------------------------------------------------------

typedef __attribute__((ext_vector_type(16))) _Float16 v16h;
typedef __attribute__((ext_vector_type(8)))  _Float16 v8h;
typedef __attribute__((ext_vector_type(8)))  float    v8f;
typedef __attribute__((ext_vector_type(4)))  int      v4i;

#define FH 128
#define NEG_SLOPE 0.2f
#define ENC_NEG_INF 0x007FFFFFu  // ordered encoding of -inf
#define BSTR 40                  // LDS B row stride (halves): 16B aligned,
                                 // 20-dword stride -> conflict-free b128 reads

#ifndef __has_builtin
#define __has_builtin(x) 0
#endif

#if __has_builtin(__builtin_amdgcn_global_load_async_to_lds_b128)
#define HAVE_ASYNC_LDS 1
#else
#define HAVE_ASYNC_LDS 0
#endif

__device__ __forceinline__ void copy_b128_to_lds(const _Float16* g, _Float16* l) {
#if HAVE_ASYNC_LDS
    // builtin signature (from toolchain diagnostic): global v4i* src,
    // lds v4i* dst, imm offset, imm cpol
    __builtin_amdgcn_global_load_async_to_lds_b128(
        (__attribute__((address_space(1))) v4i*)(_Float16*)g,
        (__attribute__((address_space(3))) v4i*)l, 0, 0);
#else
    *(v8h*)l = *(const v8h*)g;
#endif
}

__device__ __forceinline__ void wait_async_lds() {
#if HAVE_ASYNC_LDS
#if __has_builtin(__builtin_amdgcn_s_wait_asynccnt)
    __builtin_amdgcn_s_wait_asynccnt(0);
#else
    asm volatile("s_wait_asynccnt 0x0" ::: "memory");
#endif
#endif
}

__device__ __forceinline__ unsigned ordenc(float f) {
    unsigned u = __float_as_uint(f);
    return (u & 0x80000000u) ? ~u : (u | 0x80000000u);
}
__device__ __forceinline__ float orddec(unsigned e) {
    return __uint_as_float((e & 0x80000000u) ? (e & 0x7fffffffu) : ~e);
}
__device__ __forceinline__ float wred32(float v) {
#pragma unroll
    for (int o = 16; o >= 1; o >>= 1) v += __shfl_xor(v, o, 32);
    return v;
}

// ---- init: zero aggregation buffer, init per-node softmax/degree state -----
__global__ void k_init(float* agg, unsigned* menc, float* den, float* deg,
                       float* easum, int N) {
    int i = blockIdx.x * blockDim.x + threadIdx.x;
    if (i < N * FH) agg[i] = 0.0f;
    if (i < N) { menc[i] = ENC_NEG_INF; den[i] = 0.0f; deg[i] = 0.0f; easum[i] = 0.0f; }
}

// ---- c = dot(We[0,:], ae)  (one wave) --------------------------------------
__global__ void k_cdot(const float* __restrict__ We, const float* __restrict__ ae,
                       float* cbuf, int idx) {
    int lane = threadIdx.x & 31;
    float p = 0.0f;
#pragma unroll
    for (int j = 0; j < 4; ++j) p += We[lane * 4 + j] * ae[lane * 4 + j];
    p = wred32(p);
    if (lane == 0) cbuf[idx] = p;
}

// ---- degree + incoming edge-attr sum (fill_value='mean' self loops) --------
__global__ void k_deg(const int* __restrict__ dst, const float* __restrict__ ea,
                      float* deg, float* easum, int E) {
    int e = blockIdx.x * blockDim.x + threadIdx.x;
    if (e >= E) return;
    int d = dst[e];
    atomicAdd(deg + d, 1.0f);
    atomicAdd(easum + d, ea[e]);
}
__global__ void k_ealoop(float* easum, const float* __restrict__ deg, int N) {
    int n = blockIdx.x * blockDim.x + threadIdx.x;
    if (n >= N) return;
    easum[n] = easum[n] / fmaxf(deg[n], 1.0f);
}

// ---- stage W (f32 [ka,128]) -> W^T f16 [128, kp] (K zero-padded) -----------
__global__ void k_wt(const float* __restrict__ W, _Float16* __restrict__ Wt,
                     int ka, int kp) {
    int i = blockIdx.x * blockDim.x + threadIdx.x;
    if (i >= FH * kp) return;
    int n = i / kp, k = i % kp;
    Wt[(size_t)n * kp + k] = (k < ka) ? (_Float16)W[(size_t)k * FH + n] : (_Float16)0.0f;
}

// ---- stage x (f32 [N,ka]) -> f16 [N,kp] (K zero-padded, rows 16B aligned) --
__global__ void k_ah(const float* __restrict__ A, _Float16* __restrict__ Ah,
                     int N, int ka, int kp) {
    int i = blockIdx.x * blockDim.x + threadIdx.x;
    if (i >= N * kp) return;
    int r = i / kp, k = i % kp;
    Ah[i] = (k < ka) ? (_Float16)A[(size_t)r * ka + k] : (_Float16)0.0f;
}

// ---- GEMM: C[M,128] = A_f16[M,kp] x B (Wt f16 [128,kp]), wave32 WMMA -------
// 8 waves/block, one 16-row M tile per wave, all 8 N tiles per wave.
// B slice [128, 32] for the current k-chunk staged in LDS (async, double
// buffered); A fragments stream from global. Layouts per CDNA5 ISA 7.12.2.
__global__ __launch_bounds__(256) void k_gemm(const _Float16* __restrict__ A, int kp,
                                              const _Float16* __restrict__ Bt,
                                              float* __restrict__ C, int mtiles) {
    __shared__ _Float16 sB[2][FH * BSTR];

    const int tid  = threadIdx.x;
    const int wid  = (blockIdx.x * blockDim.x + tid) >> 5;
    const int lane = tid & 31;
    const int mlo  = lane & 15;
    const int hi   = lane >> 4;            // 0: lanes 0-15, 1: lanes 16-31
    const bool active = (wid < mtiles);    // wave-uniform (EXEC all-ones inside)

    const _Float16* Arow =
        A + (size_t)((active ? wid : 0) * 16 + mlo) * (size_t)kp;

    v8f acc[8];
#pragma unroll
    for (int nt = 0; nt < 8; ++nt) acc[nt] = (v8f)0.0f;

    // stage chunk 0: 128 rows x 32 halves = 512 b128 transfers, 2 per thread
#pragma unroll
    for (int r = 0; r < 2; ++r) {
        int idx = tid + 256 * r;
        int n = idx >> 2, p = idx & 3;
        copy_b128_to_lds(Bt + (size_t)n * kp + p * 8, &sB[0][n * BSTR + p * 8]);
    }
    wait_async_lds();
    __syncthreads();

    const int nsteps = kp >> 5;
    for (int step = 0; step < nsteps; ++step) {
        const int k0  = step << 5;
        const int cur = step & 1;
        const int nxt = cur ^ 1;

        if (step + 1 < nsteps) {           // prefetch next B slice into LDS
#pragma unroll
            for (int r = 0; r < 2; ++r) {
                int idx = tid + 256 * r;
                int n = idx >> 2, p = idx & 3;
                copy_b128_to_lds(Bt + (size_t)n * kp + (k0 + 32) + p * 8,
                                 &sB[nxt][n * BSTR + p * 8]);
            }
        }

        if (active) {
            // A fragment: lanes0-15 K{k0..+7, k0+16..+23}; lanes16-31 +8
            v8h alo = *(const v8h*)(Arow + k0 + hi * 8);
            v8h ahi = *(const v8h*)(Arow + k0 + 16 + hi * 8);
            v16h afrag = __builtin_shufflevector(alo, ahi,
                0, 1, 2, 3, 4, 5, 6, 7, 8, 9, 10, 11, 12, 13, 14, 15);
#pragma unroll
            for (int nt = 0; nt < 8; ++nt) {
                // B fragment from LDS: lanes0-15 -> local K {0..7, 16..23}; lanes16-31 +8
                const _Float16* bp = &sB[cur][(nt * 16 + mlo) * BSTR + hi * 8];
                v8h blo = *(const v8h*)bp;
                v8h bhi = *(const v8h*)(bp + 16);
                v16h bfrag = __builtin_shufflevector(blo, bhi,
                    0, 1, 2, 3, 4, 5, 6, 7, 8, 9, 10, 11, 12, 13, 14, 15);
                acc[nt] = __builtin_amdgcn_wmma_f32_16x16x32_f16(
                    false, afrag, false, bfrag, (short)0, acc[nt], false, false);
            }
        }

        wait_async_lds();                  // next slice resident
        __syncthreads();                   // all waves done reading cur
    }

    if (active) {
        // D layout: VGPR r -> M = r (lanes 0-15) / 8+r (lanes 16-31), N = lane&15
        const int row0 = wid * 16 + hi * 8;
#pragma unroll
        for (int nt = 0; nt < 8; ++nt)
#pragma unroll
            for (int r = 0; r < 8; ++r)
                C[(size_t)(row0 + r) * FH + nt * 16 + mlo] = acc[nt][r];
    }
}

// ---- per-node attention dots: a_s[n]=h[n].as, a_d[n]=h[n].ad (wave/node) ---
__global__ void k_dot2(const float* __restrict__ h, const float* __restrict__ asv,
                       const float* __restrict__ adv, float* a_s, float* a_d, int N) {
    int wid = (blockIdx.x * blockDim.x + threadIdx.x) >> 5;
    if (wid >= N) return;
    int lane = threadIdx.x & 31;
    float4 hv = *(const float4*)(h + (size_t)wid * FH + lane * 4);
    float4 sv = *(const float4*)(asv + lane * 4);
    float4 dv = *(const float4*)(adv + lane * 4);
    float ps = hv.x * sv.x + hv.y * sv.y + hv.z * sv.z + hv.w * sv.w;
    float pd = hv.x * dv.x + hv.y * dv.y + hv.z * dv.z + hv.w * dv.w;
    ps = wred32(ps); pd = wred32(pd);
    if (lane == 0) { a_s[wid] = ps; a_d[wid] = pd; }
}

// ---- alpha = leaky_relu(a_s[s]+a_d[d]+c*ea) and segment max (edges+loops) --
__global__ void k_alpha_max(const int* __restrict__ src, const int* __restrict__ dst,
                            const float* __restrict__ ea, const float* __restrict__ eal,
                            const float* __restrict__ a_s, const float* __restrict__ a_d,
                            const float* __restrict__ cbuf, int cidx,
                            float* alpha, unsigned* menc, int E, int N) {
    int i = blockIdx.x * blockDim.x + threadIdx.x;
    if (i >= E + N) return;
    float c = cbuf[cidx];
    int s, d; float e;
    if (i < E) { s = src[i]; d = dst[i]; e = ea[i]; }
    else       { s = d = i - E;          e = eal[s]; }
    float v = a_s[s] + a_d[d] + c * e;
    v = (v > 0.0f) ? v : NEG_SLOPE * v;
    alpha[i] = v;
    atomicMax(menc + d, ordenc(v));
}

// ---- ex = exp(alpha - m[d]); den[d] += ex (in-place over alpha) ------------
__global__ void k_expsum(const int* __restrict__ dst, float* alpha,
                         const unsigned* __restrict__ menc, float* den, int E, int N) {
    int i = blockIdx.x * blockDim.x + threadIdx.x;
    if (i >= E + N) return;
    int d = (i < E) ? dst[i] : (i - E);
    float ex = __expf(alpha[i] - orddec(menc[d]));
    alpha[i] = ex;
    atomicAdd(den + d, ex);
}

// ---- aggregation: agg[d] += (ex/den[d]) * h[s]  (one wave per message) -----
__global__ __launch_bounds__(256) void k_agg(const int* __restrict__ src,
                                             const int* __restrict__ dst,
                                             const float* __restrict__ ex,
                                             const float* __restrict__ den,
                                             const float* __restrict__ h,
                                             float* agg, int E, int N) {
    int wid = (blockIdx.x * blockDim.x + threadIdx.x) >> 5;
    if (wid >= E + N) return;
    int lane = threadIdx.x & 31;
    int s, d;
    if (wid < E) { s = src[wid]; d = dst[wid]; }
    else         { s = d = wid - E; }
    float coef = ex[wid] / den[d];
    float4 hv = *(const float4*)(h + (size_t)s * FH + lane * 4);
    float* o = agg + (size_t)d * FH + lane * 4;
    atomicAdd(o + 0, coef * hv.x);
    atomicAdd(o + 1, coef * hv.y);
    atomicAdd(o + 2, coef * hv.z);
    atomicAdd(o + 3, coef * hv.w);
}

// ---- out = relu(agg + b); emit f16 copy; reset agg/m/den for next layer ----
__global__ void k_biasrelu(float* agg, const float* __restrict__ b,
                           float* hR, _Float16* hRh,
                           unsigned* menc, float* den, int N) {
    int i = blockIdx.x * blockDim.x + threadIdx.x;
    if (i >= N * FH) return;
    int col = i & (FH - 1);
    float v = agg[i] + b[col];
    v = fmaxf(v, 0.0f);
    hR[i] = v;
    hRh[i] = (_Float16)v;
    agg[i] = 0.0f;
    if (col == 0) { menc[i >> 7] = ENC_NEG_INF; den[i >> 7] = 0.0f; }
}

// ---- node head: b_out[n] = (h[n].Wb + bb) * mask[n] ------------------------
__global__ void k_node_head(const float* __restrict__ h, const float* __restrict__ Wb,
                            const float* __restrict__ bb, const float* __restrict__ mask,
                            float* outb, int N) {
    int wid = (blockIdx.x * blockDim.x + threadIdx.x) >> 5;
    if (wid >= N) return;
    int lane = threadIdx.x & 31;
    float4 hv = *(const float4*)(h + (size_t)wid * FH + lane * 4);
    float4 wv = *(const float4*)(Wb + lane * 4);
    float p = hv.x * wv.x + hv.y * wv.y + hv.z * wv.z + hv.w * wv.w;
    p = wred32(p);
    if (lane == 0) outb[wid] = (p + bb[0]) * mask[wid];
}

// ---- edge head: w[e] = 0.5*(h[s]+h[d]).Ww + bw -----------------------------
__global__ void k_edge_head(const float* __restrict__ h, const int* __restrict__ src,
                            const int* __restrict__ dst, const float* __restrict__ Ww,
                            const float* __restrict__ bw, float* outw, int E) {
    int wid = (blockIdx.x * blockDim.x + threadIdx.x) >> 5;
    if (wid >= E) return;
    int lane = threadIdx.x & 31;
    int s = src[wid], d = dst[wid];
    float4 a = *(const float4*)(h + (size_t)s * FH + lane * 4);
    float4 b = *(const float4*)(h + (size_t)d * FH + lane * 4);
    float4 wv = *(const float4*)(Ww + lane * 4);
    float p = 0.5f * ((a.x + b.x) * wv.x + (a.y + b.y) * wv.y +
                      (a.z + b.z) * wv.z + (a.w + b.w) * wv.w);
    p = wred32(p);
    if (lane == 0) outw[wid] = p + bw[0];
}

// ---------------------------------------------------------------------------
extern "C" void kernel_launch(void* const* d_in, const int* in_sizes, int n_in,
                              void* d_out, int out_size, void* d_ws, size_t ws_size,
                              hipStream_t stream) {
    (void)n_in; (void)out_size; (void)ws_size;

    const float* x    = (const float*)d_in[0];
    const int*   ei   = (const int*)d_in[1];
    const float* ea   = (const float*)d_in[2];
    const float* mask = (const float*)d_in[3];
    const float* W1   = (const float*)d_in[4];
    const float* b1   = (const float*)d_in[5];
    const float* as1  = (const float*)d_in[6];
    const float* ad1  = (const float*)d_in[7];
    const float* We1  = (const float*)d_in[8];
    const float* ae1  = (const float*)d_in[9];
    const float* W2   = (const float*)d_in[10];
    const float* b2   = (const float*)d_in[11];
    const float* as2  = (const float*)d_in[12];
    const float* ad2  = (const float*)d_in[13];
    const float* We2  = (const float*)d_in[14];
    const float* ae2  = (const float*)d_in[15];
    const float* Wb   = (const float*)d_in[16];
    const float* bb   = (const float*)d_in[17];
    const float* Ww   = (const float*)d_in[18];
    const float* bw   = (const float*)d_in[19];
    float* out = (float*)d_out;

    const int N   = in_sizes[3];            // 50000 (multiple of 16)
    const int E   = in_sizes[1] / 2;        // 1.6M
    const int FIN = in_sizes[0] / N;        // 503
    const int KP1 = (FIN + 31) & ~31;       // 512
    const int mtiles = N / 16;

    const int* src = ei;
    const int* dst = ei + E;

    // workspace carve-up (256B aligned)
    char* w = (char*)d_ws;
    size_t off = 0;
    auto take = [&](size_t bytes) -> void* {
        void* p = w + off;
        off = (off + bytes + 255) & ~(size_t)255;
        return p;
    };
    _Float16* xh   = (_Float16*)take((size_t)N * KP1 * 2);
    _Float16* w1t  = (_Float16*)take((size_t)FH * KP1 * 2);
    _Float16* w2t  = (_Float16*)take((size_t)FH * FH * 2);
    float*    hA   = (float*)take((size_t)N * FH * 4);     // GEMM output h
    float*    agg  = (float*)take((size_t)N * FH * 4);     // aggregation accum
    float*    hR   = (float*)take((size_t)N * FH * 4);     // relu(agg+b)
    _Float16* hRh  = (_Float16*)take((size_t)N * FH * 2);  // f16 copy for GEMM2
    float*    a_s  = (float*)take((size_t)N * 4);
    float*    a_d  = (float*)take((size_t)N * 4);
    float*    deg  = (float*)take((size_t)N * 4);
    float*    eal  = (float*)take((size_t)N * 4);          // easum -> ea_loop
    unsigned* menc = (unsigned*)take((size_t)N * 4);
    float*    den  = (float*)take((size_t)N * 4);
    float*    alph = (float*)take((size_t)(E + N) * 4);    // alpha -> ex
    float*    cbuf = (float*)take(2 * 4);

    const int T = 256;
    const int gNF  = (N * FH + T - 1) / T;
    const int gN   = (N + T - 1) / T;
    const int gE   = (E + T - 1) / T;
    const int gEN  = (E + N + T - 1) / T;
    const int gENw = (E + N + 7) / 8;       // one wave per message
    const int gNw  = (N + 7) / 8;           // one wave per node
    const int gEw  = (E + 7) / 8;           // one wave per edge
    const int gMT  = (mtiles + 7) / 8;      // 8 waves (M tiles) per block

    // ---- shared preprocessing ----
    k_init<<<gNF, T, 0, stream>>>(agg, menc, den, deg, eal, N);
    k_cdot<<<1, 32, 0, stream>>>(We1, ae1, cbuf, 0);
    k_cdot<<<1, 32, 0, stream>>>(We2, ae2, cbuf, 1);
    k_deg<<<gE, T, 0, stream>>>(dst, ea, deg, eal, E);
    k_ealoop<<<gN, T, 0, stream>>>(eal, deg, N);
    k_wt<<<(FH * KP1 + T - 1) / T, T, 0, stream>>>(W1, w1t, FIN, KP1);
    k_wt<<<(FH * FH + T - 1) / T, T, 0, stream>>>(W2, w2t, FH, FH);
    k_ah<<<(N * KP1 + T - 1) / T, T, 0, stream>>>(x, xh, N, FIN, KP1);

    // ---- layer 1 ----
    k_gemm<<<gMT, T, 0, stream>>>(xh, KP1, w1t, hA, mtiles);
    k_dot2<<<gNw, T, 0, stream>>>(hA, as1, ad1, a_s, a_d, N);
    k_alpha_max<<<gEN, T, 0, stream>>>(src, dst, ea, eal, a_s, a_d, cbuf, 0, alph, menc, E, N);
    k_expsum<<<gEN, T, 0, stream>>>(dst, alph, menc, den, E, N);
    k_agg<<<gENw, T, 0, stream>>>(src, dst, alph, den, hA, agg, E, N);
    k_biasrelu<<<gNF, T, 0, stream>>>(agg, b1, hR, hRh, menc, den, N);

    // ---- layer 2 ----
    k_gemm<<<gMT, T, 0, stream>>>(hRh, FH, w2t, hA, mtiles);
    k_dot2<<<gNw, T, 0, stream>>>(hA, as2, ad2, a_s, a_d, N);
    k_alpha_max<<<gEN, T, 0, stream>>>(src, dst, ea, eal, a_s, a_d, cbuf, 1, alph, menc, E, N);
    k_expsum<<<gEN, T, 0, stream>>>(dst, alph, menc, den, E, N);
    k_agg<<<gENw, T, 0, stream>>>(src, dst, alph, den, hA, agg, E, N);
    k_biasrelu<<<gNF, T, 0, stream>>>(agg, b2, hR, hRh, menc, den, N);

    // ---- heads: d_out = [ w (E) | b_out (N) ] ----
    k_edge_head<<<gEw, T, 0, stream>>>(hR, src, dst, Ww, bw, out, E);
    k_node_head<<<gNw, T, 0, stream>>>(hR, Wb, bb, mask, out + E, N);
}
